// my_Net_37589553774844
// MI455X (gfx1250) — matrix operand order repaired and checked by reading, not compiled, not run on hardware
//
#include <hip/hip_runtime.h>
#include <hip/hip_bf16.h>

#define NNODES 6144
#define FIN    1000
#define KPAD   1024
#define HID    256
#define EMB    16
#define KCLUS  10
#define ALPHA  0.2f

typedef __attribute__((ext_vector_type(16))) __bf16 v16bf;
typedef __attribute__((ext_vector_type(8)))  float  v8f;

union BF16Frag { v16bf v; unsigned short u[16]; uint4 q[2]; };

__device__ __forceinline__ unsigned short f2bfu(float f) {
  unsigned u = __builtin_bit_cast(unsigned, f);
  u += 0x7FFFu + ((u >> 16) & 1u);          // round-to-nearest-even
  return (unsigned short)(u >> 16);
}

// ---------------- prep kernels ----------------

__global__ void k_cvt_pad(const float* __restrict__ src, unsigned short* __restrict__ dst,
                          int srows, int scols, int drows, int dcols) {
  int idx = blockIdx.x * blockDim.x + threadIdx.x;
  if (idx >= drows * dcols) return;
  int r = idx / dcols, c = idx % dcols;
  float v = (r < srows && c < scols) ? src[(size_t)r * scols + c] : 0.0f;
  dst[idx] = f2bfu(v);
}

__global__ void k_colsum(const float* __restrict__ adj, float* __restrict__ deg_inv, int N) {
  int j = blockIdx.x * blockDim.x + threadIdx.x;
  if (j >= N) return;
  float sum = 0.0f;
  for (int i = 0; i < N; ++i) sum += adj[(size_t)i * N + j];
  deg_inv[j] = 1.0f / sum;
}

// ---------------- generic bf16 WMMA GEMM: C[M,N] = A[M,K] @ B[K,N] ----------------
// block = 128 threads = 4 waves; wave w -> 16x16 tile at (blockIdx.x*64 + w*16, blockIdx.y*16)
// used for X@W1, h1@W2 and sigmoid(z@zT) (N too large there for N-strip blocking)

__global__ __launch_bounds__(128)
void k_gemm_bf16(const unsigned short* __restrict__ A, int lda,
                 const unsigned short* __restrict__ B, int ldb,
                 float* __restrict__ Cf, int ldc,
                 unsigned short* __restrict__ Cbf, int ldcbf,
                 int K, int applySigmoid) {
  const int lane = threadIdx.x & 31;
  const int wave = threadIdx.x >> 5;
  const int m0 = blockIdx.x * 64 + wave * 16;
  const int n0 = blockIdx.y * 16;
  const int arow = m0 + (lane & 15);
  const int kb = (lane < 16) ? 0 : 8;

  v8f acc = {0.f, 0.f, 0.f, 0.f, 0.f, 0.f, 0.f, 0.f};
  for (int k0 = 0; k0 < K; k0 += 32) {
    BF16Frag a, b;
    const unsigned short* ap = A + (size_t)arow * lda + k0 + kb;
    a.q[0] = *reinterpret_cast<const uint4*>(ap);
    a.q[1] = *reinterpret_cast<const uint4*>(ap + 16);
    const unsigned short* bp = B + (size_t)(k0 + kb) * ldb + n0 + (lane & 15);
#pragma unroll
    for (int u = 0; u < 8; ++u) {
      b.u[u]     = bp[(size_t)u * ldb];
      b.u[u + 8] = bp[(size_t)(16 + u) * ldb];
    }
    acc = __builtin_amdgcn_wmma_f32_16x16x32_bf16(false, a.v, false, b.v,
                                                  (short)0, acc, false, false);
  }
  const int col = n0 + (lane & 15);
  const int rbase = m0 + ((lane >> 4) << 3);
#pragma unroll
  for (int r = 0; r < 8; ++r) {
    int row = rbase + r;
    float vf = acc[r];
    if (applySigmoid) vf = 1.0f / (1.0f + __expf(-vf));
    Cf[(size_t)row * ldc + col] = vf;
    if (Cbf) Cbf[(size_t)row * ldcbf + col] = f2bfu(vf);
  }
}

// ---------------- WHE = diag(deg_inv) @ adj^T @ Hbf ----------------
// N-strip blocked: each wave computes 16 rows x NT*16 cols; adj is scanned exactly once,
// A fragment reused across NT back-to-back WMMAs.

template <int NT>
__global__ __launch_bounds__(128)
void k_gemm_adjT(const float* __restrict__ adj,
                 const unsigned short* __restrict__ B,        // [N, NT*16] bf16
                 const float* __restrict__ deg_inv,
                 float* __restrict__ Cf, unsigned short* __restrict__ Cbf, int N) {
  const int lane = threadIdx.x & 31;
  const int wave = threadIdx.x >> 5;
  const int m0 = blockIdx.x * 64 + wave * 16;
  const int m = m0 + (lane & 15);
  const int kb = (lane < 16) ? 0 : 8;
  const int ldb = NT * 16;

  v8f acc[NT];
#pragma unroll
  for (int nt = 0; nt < NT; ++nt) acc[nt] = (v8f){0.f,0.f,0.f,0.f,0.f,0.f,0.f,0.f};

  for (int k0 = 0; k0 < N; k0 += 32) {
    BF16Frag a;
#pragma unroll
    for (int u = 0; u < 8; ++u) {
      a.u[u]     = f2bfu(adj[(size_t)(k0 + kb + u)      * N + m]);   // A[m][k] = adj[k][m]
      a.u[u + 8] = f2bfu(adj[(size_t)(k0 + kb + 16 + u) * N + m]);
    }
    const unsigned short* brow = B + (size_t)(k0 + kb) * ldb + (lane & 15);
#pragma unroll
    for (int nt = 0; nt < NT; ++nt) {
      BF16Frag b;
#pragma unroll
      for (int u = 0; u < 8; ++u) {
        b.u[u]     = brow[(size_t)u * ldb + nt * 16];
        b.u[u + 8] = brow[(size_t)(16 + u) * ldb + nt * 16];
      }
      acc[nt] = __builtin_amdgcn_wmma_f32_16x16x32_bf16(false, a.v, false, b.v,
                                                        (short)0, acc[nt], false, false);
    }
  }
  const int rbase = m0 + ((lane >> 4) << 3);
#pragma unroll
  for (int nt = 0; nt < NT; ++nt) {
    const int col = nt * 16 + (lane & 15);
#pragma unroll
    for (int r = 0; r < 8; ++r) {
      int row = rbase + r;
      float vf = acc[nt][r] * deg_inv[row];
      Cf[(size_t)row * ldb + col] = vf;
      Cbf[(size_t)row * ldb + col] = f2bfu(vf);
    }
  }
}

// ---------------- GEMV: out[i] = sum_c H[i][c] * a[c]  (one wave per row) ----------------

__global__ void k_gemv(const float* __restrict__ H, int ldh, int C,
                       const float* __restrict__ a, float* __restrict__ out, int N) {
  int lane = threadIdx.x & 31;
  int row = blockIdx.x * (blockDim.x >> 5) + (threadIdx.x >> 5);
  if (row >= N) return;
  float sum = 0.0f;
  for (int c = lane; c < C; c += 32) sum += H[(size_t)row * ldh + c] * a[c];
#pragma unroll
  for (int off = 16; off > 0; off >>= 1) sum += __shfl_down(sum, off);
  if (lane == 0) out[row] = sum;
}

// ---------------- per-row masked-softmax stats: m_i = max, Z_i = sum exp(e - m) ----------------

__global__ void k_rowstats(const float* __restrict__ adj,
                           const float* __restrict__ s, const float* __restrict__ t,
                           float* __restrict__ mout, float* __restrict__ zout, int N) {
  __shared__ float sm[256], sz[256];
  const int i = blockIdx.x;
  const int tid = threadIdx.x;
  const float si = s[i];
  float m = -3.0e38f, Z = 0.0f;
  for (int j = tid; j < N; j += 256) {
    if (adj[(size_t)i * N + j] > 0.0f) {
      float e = si + t[j];
      e = (e >= 0.0f) ? e : ALPHA * e;               // LeakyReLU
      if (e > m) { Z = Z * __expf(m - e) + 1.0f; m = e; }
      else         Z += __expf(e - m);
    }
  }
  sm[tid] = m; sz[tid] = Z;
  __syncthreads();
  for (int off = 128; off > 0; off >>= 1) {
    if (tid < off) {
      float m1 = sm[tid], z1 = sz[tid], m2 = sm[tid + off], z2 = sz[tid + off];
      float mhi = fmaxf(m1, m2), mlo = fminf(m1, m2);
      float zhi = (m1 >= m2) ? z1 : z2, zlo = (m1 >= m2) ? z2 : z1;
      sm[tid] = mhi;
      sz[tid] = zhi + ((mlo > -1.0e38f) ? zlo * __expf(mlo - mhi) : 0.0f);
    }
    __syncthreads();
  }
  if (tid == 0) {
    float mf = sm[0], zf = sz[0];
    if (mf < -1.0e38f) { mf = 0.0f; zf = 1.0f; }     // no neighbors (doesn't occur in practice)
    mout[i] = mf;
    zout[i] = fmaxf(zf, 1e-30f);
  }
}

// ---------------- fused attention GEMM: out = 0.5 * softmax(mask(lrelu(s_i+t_j))) @ WHE + 0.5 * H ----------------
// N-strip blocked like k_gemm_adjT: adj scanned once, exp/mask/cvt computed once per element,
// A fragment reused across NT consecutive WMMAs.

template <int NT>
__global__ __launch_bounds__(128)
void k_gemm_attn(const float* __restrict__ adj,
                 const float* __restrict__ s, const float* __restrict__ t,
                 const float* __restrict__ mrow, const float* __restrict__ Zrow,
                 const unsigned short* __restrict__ B,        // WHE bf16 [N, NT*16]
                 const float* __restrict__ Hf,                // H f32 [N, NT*16]
                 float* __restrict__ Cf, unsigned short* __restrict__ Cbf, int N) {
  const int lane = threadIdx.x & 31;
  const int wave = threadIdx.x >> 5;
  const int m0 = blockIdx.x * 64 + wave * 16;
  const int i = m0 + (lane & 15);
  const int kb = (lane < 16) ? 0 : 8;
  const int ldb = NT * 16;
  const float si = s[i];
  const float mi = mrow[i];
  const float* arowp = adj + (size_t)i * N;

  v8f acc[NT];
#pragma unroll
  for (int nt = 0; nt < NT; ++nt) acc[nt] = (v8f){0.f,0.f,0.f,0.f,0.f,0.f,0.f,0.f};

  for (int k0 = 0; k0 < N; k0 += 32) {
    __builtin_prefetch(arowp + k0 + 1024, 0, 1);     // prefetch same adj row i, 1024 floats ahead -> global_prefetch_b8
    float av[16], tv[16];
    const float* ap = arowp + k0 + kb;
    const float* tp = t + k0 + kb;
    *(float4*)(av + 0)  = *(const float4*)(ap + 0);
    *(float4*)(av + 4)  = *(const float4*)(ap + 4);
    *(float4*)(av + 8)  = *(const float4*)(ap + 16);
    *(float4*)(av + 12) = *(const float4*)(ap + 20);
    *(float4*)(tv + 0)  = *(const float4*)(tp + 0);
    *(float4*)(tv + 4)  = *(const float4*)(tp + 4);
    *(float4*)(tv + 8)  = *(const float4*)(tp + 16);
    *(float4*)(tv + 12) = *(const float4*)(tp + 20);
    BF16Frag a;
#pragma unroll
    for (int u = 0; u < 16; ++u) {
      float e = si + tv[u];
      e = (e >= 0.0f) ? e : ALPHA * e;
      float p = (av[u] > 0.0f) ? __expf(e - mi) : 0.0f;
      a.u[u] = f2bfu(p);
    }
    const unsigned short* brow = B + (size_t)(k0 + kb) * ldb + (lane & 15);
#pragma unroll
    for (int nt = 0; nt < NT; ++nt) {
      BF16Frag b;
#pragma unroll
      for (int u = 0; u < 8; ++u) {
        b.u[u]     = brow[(size_t)u * ldb + nt * 16];
        b.u[u + 8] = brow[(size_t)(16 + u) * ldb + nt * 16];
      }
      acc[nt] = __builtin_amdgcn_wmma_f32_16x16x32_bf16(false, a.v, false, b.v,
                                                        (short)0, acc[nt], false, false);
    }
  }
  const int rbase = m0 + ((lane >> 4) << 3);
#pragma unroll
  for (int nt = 0; nt < NT; ++nt) {
    const int col = nt * 16 + (lane & 15);
#pragma unroll
    for (int r = 0; r < 8; ++r) {
      int row = rbase + r;
      float vf = acc[nt][r] * (0.5f / Zrow[row]) + 0.5f * Hf[(size_t)row * ldb + col];
      Cf[(size_t)row * ldb + col] = vf;
      if (Cbf) Cbf[(size_t)row * ldb + col] = f2bfu(vf);
    }
  }
}

// ---------------- z = row-normalize(h2); build padded bf16 z [N,32] and z^T [32,N] ----------------

__global__ void k_znorm(const float* __restrict__ h, float* __restrict__ zf,
                        unsigned short* __restrict__ zbf, unsigned short* __restrict__ ztbf, int N) {
  int tid = threadIdx.x;
  int col = tid & 15;
  int row = blockIdx.x * 16 + (tid >> 4);
  float v = h[(size_t)row * EMB + col];
  float sq = v * v;
#pragma unroll
  for (int off = 8; off > 0; off >>= 1) sq += __shfl_xor(sq, off, 16);
  float z = v * rsqrtf(fmaxf(sq, 1e-30f));
  zf[(size_t)row * EMB + col] = z;
  unsigned short zb = f2bfu(z);
  zbf[(size_t)row * 32 + col] = zb;
  zbf[(size_t)row * 32 + 16 + col] = 0;              // K padding 16 -> 32
  ztbf[(size_t)col * N + row] = zb;
  ztbf[(size_t)(col + 16) * N + row] = 0;
}

// ---------------- student-t cluster assignment (V=1 => exponent 1) ----------------

__global__ void k_q(const float* __restrict__ z, const float* __restrict__ cl,
                    float* __restrict__ q, int N) {
  int i = blockIdx.x * blockDim.x + threadIdx.x;
  if (i >= N) return;
  float zr[EMB];
#pragma unroll
  for (int c = 0; c < EMB; ++c) zr[c] = z[(size_t)i * EMB + c];
  float qv[KCLUS], sumq = 0.0f;
#pragma unroll
  for (int k = 0; k < KCLUS; ++k) {
    float d2 = 0.0f;
#pragma unroll
    for (int c = 0; c < EMB; ++c) { float d = zr[c] - cl[k * EMB + c]; d2 += d * d; }
    float tt = 1.0f / (1.0f + d2);
    qv[k] = tt; sumq += tt;
  }
  float inv = 1.0f / sumq;
#pragma unroll
  for (int k = 0; k < KCLUS; ++k) q[(size_t)i * KCLUS + k] = qv[k] * inv;
}

// ---------------- launch ----------------

extern "C" void kernel_launch(void* const* d_in, const int* in_sizes, int n_in,
                              void* d_out, int out_size, void* d_ws, size_t ws_size,
                              hipStream_t stream) {
  const float* x         = (const float*)d_in[0];
  const float* adj       = (const float*)d_in[1];
  const float* W1        = (const float*)d_in[2];
  const float* a_self1   = (const float*)d_in[3];
  const float* a_neighs1 = (const float*)d_in[4];
  const float* W2        = (const float*)d_in[5];
  const float* a_self2   = (const float*)d_in[6];
  const float* a_neighs2 = (const float*)d_in[7];
  const float* clusters  = (const float*)d_in[8];
  float* out = (float*)d_out;
  const int N = NNODES;

  char* wp = (char*)d_ws;
  auto alloc = [&](size_t bytes) -> char* {
    char* p = wp; wp += (bytes + 255) & ~(size_t)255; return p;
  };
  unsigned short* Xbf   = (unsigned short*)alloc((size_t)N * KPAD * 2);
  unsigned short* W1bf  = (unsigned short*)alloc((size_t)KPAD * HID * 2);
  unsigned short* W2bf  = (unsigned short*)alloc((size_t)HID * EMB * 2);
  float*          degi  = (float*)alloc((size_t)N * 4);
  float*          H1f   = (float*)alloc((size_t)N * HID * 4);
  unsigned short* H1bf  = (unsigned short*)alloc((size_t)N * HID * 2);
  float*          WHE1f = (float*)alloc((size_t)N * HID * 4);
  unsigned short* WHE1b = (unsigned short*)alloc((size_t)N * HID * 2);
  float* s1 = (float*)alloc((size_t)N * 4);
  float* t1 = (float*)alloc((size_t)N * 4);
  float* m1 = (float*)alloc((size_t)N * 4);
  float* Z1 = (float*)alloc((size_t)N * 4);
  float*          h1f   = (float*)alloc((size_t)N * HID * 4);
  unsigned short* h1bf  = (unsigned short*)alloc((size_t)N * HID * 2);
  float*          H2f   = (float*)alloc((size_t)N * EMB * 4);
  unsigned short* H2bf  = (unsigned short*)alloc((size_t)N * EMB * 2);
  float*          WHE2f = (float*)alloc((size_t)N * EMB * 4);
  unsigned short* WHE2b = (unsigned short*)alloc((size_t)N * EMB * 2);
  float* s2 = (float*)alloc((size_t)N * 4);
  float* t2 = (float*)alloc((size_t)N * 4);
  float* m2 = (float*)alloc((size_t)N * 4);
  float* Z2 = (float*)alloc((size_t)N * 4);
  float*          h2f   = (float*)alloc((size_t)N * EMB * 4);
  unsigned short* zbf   = (unsigned short*)alloc((size_t)N * 32 * 2);
  unsigned short* ztbf  = (unsigned short*)alloc((size_t)32 * N * 2);

  float* zf   = out + (size_t)N * N;
  float* qout = out + (size_t)N * N + (size_t)N * EMB;

  // prep
  k_cvt_pad<<<(N * KPAD + 255) / 256, 256, 0, stream>>>(x, Xbf, N, FIN, N, KPAD);
  k_cvt_pad<<<(KPAD * HID + 255) / 256, 256, 0, stream>>>(W1, W1bf, FIN, HID, KPAD, HID);
  k_cvt_pad<<<(HID * EMB + 255) / 256, 256, 0, stream>>>(W2, W2bf, HID, EMB, HID, EMB);
  k_colsum<<<N / 256, 256, 0, stream>>>(adj, degi, N);

  // layer 1
  k_gemm_bf16<<<dim3(N / 64, HID / 16), 128, 0, stream>>>(Xbf, KPAD, W1bf, HID,
                                                          H1f, HID, H1bf, HID, KPAD, 0);
  k_gemm_adjT<16><<<N / 64, 128, 0, stream>>>(adj, H1bf, degi, WHE1f, WHE1b, N);
  k_gemv<<<N / 8, 256, 0, stream>>>(H1f, HID, HID, a_self1, s1, N);
  k_gemv<<<N / 8, 256, 0, stream>>>(WHE1f, HID, HID, a_neighs1, t1, N);
  k_rowstats<<<N, 256, 0, stream>>>(adj, s1, t1, m1, Z1, N);
  k_gemm_attn<16><<<N / 64, 128, 0, stream>>>(adj, s1, t1, m1, Z1,
                                              WHE1b, H1f, h1f, h1bf, N);
  // layer 2
  k_gemm_bf16<<<dim3(N / 64, EMB / 16), 128, 0, stream>>>(h1bf, HID, W2bf, EMB,
                                                          H2f, EMB, H2bf, EMB, HID, 0);
  k_gemm_adjT<1><<<N / 64, 128, 0, stream>>>(adj, H2bf, degi, WHE2f, WHE2b, N);
  k_gemv<<<N / 8, 256, 0, stream>>>(H2f, EMB, EMB, a_self2, s2, N);
  k_gemv<<<N / 8, 256, 0, stream>>>(WHE2f, EMB, EMB, a_neighs2, t2, N);
  k_rowstats<<<N, 256, 0, stream>>>(adj, s2, t2, m2, Z2, N);
  k_gemm_attn<1><<<N / 64, 128, 0, stream>>>(adj, s2, t2, m2, Z2,
                                             WHE2b, H2f, h2f, nullptr, N);
  // outputs
  k_znorm<<<N / 16, 256, 0, stream>>>(h2f, zf, zbf, ztbf, N);
  k_q<<<N / 256, 256, 0, stream>>>(zf, clusters, qout, N);
  k_gemm_bf16<<<dim3(N / 64, N / 16), 128, 0, stream>>>(zbf, 32, ztbf, N,
                                                        out, N, nullptr, 0, 32, 1);
}
